// Attention_71966472012244
// MI455X (gfx1250) — matrix operand order repaired and checked by reading, not compiled, not run on hardware
//
#include <hip/hip_runtime.h>
#include <hip/hip_bf16.h>

typedef __attribute__((ext_vector_type(16))) __bf16 v16bf;
typedef __attribute__((ext_vector_type(8)))  float  v8f;

static constexpr int Bc = 256;   // batch
static constexpr int Tc = 256;   // time (attention length)
static constexpr int INc = 512;  // encoder feature dim
static constexpr int Hc = 512;   // hidden
static constexpr int NCc = 97;   // num classes
static constexpr int Ec = 256;   // embedding dim
static constexpr int STEPS = 25;

// ---------------------------------------------------------------------------
// helpers
// ---------------------------------------------------------------------------
__device__ __forceinline__ float fast_tanh(float x) {
  float e = __expf(2.0f * x);
  return 1.0f - 2.0f / (e + 1.0f);
}
__device__ __forceinline__ float sigmoidf(float x) {
  return 1.0f / (1.0f + __expf(-x));
}

// A-fragment (16x32 bf16, M x K): lane m = lane&15, half = lane>>4.
// Lane holds K = half*8 + {0..7} and half*8+16 + {0..7}  -> two b128 loads.
__device__ __forceinline__ void load_a_frag(const __bf16* A, int K, int row,
                                            int k0, int half, v16bf& a) {
  const __bf16* p = A + (size_t)row * K + k0 + half * 8;
  uint4* u = reinterpret_cast<uint4*>(&a);
  u[0] = *reinterpret_cast<const uint4*>(p);
  u[1] = *reinterpret_cast<const uint4*>(p + 16);
}

// B-fragment (32x16 bf16, K x N) from row-major W[N,K] (B[k][n] = W[n][k]):
// lane n = lane&15 holds K = half*8 + {0..7} and half*8+16 + {0..7}  -> two b128 loads.
__device__ __forceinline__ void load_b_frag(const __bf16* W, int K, int col,
                                            int k0, int half, v16bf& b) {
  const __bf16* p = W + (size_t)col * K + k0 + half * 8;
  uint4* u = reinterpret_cast<uint4*>(&b);
  u[0] = *reinterpret_cast<const uint4*>(p);
  u[1] = *reinterpret_cast<const uint4*>(p + 16);
}

__device__ __forceinline__ void gemm_k_loop(const __bf16* A, const __bf16* W,
                                            int K, int N, int rowBase,
                                            int colBase, int half, int l15,
                                            v8f acc[2][4]) {
  for (int k0 = 0; k0 < K; k0 += 32) {
    v16bf a[2], b[4];
#pragma unroll
    for (int t = 0; t < 2; ++t)
      load_a_frag(A, K, rowBase + t * 16 + l15, k0, half, a[t]);
#pragma unroll
    for (int u = 0; u < 4; ++u) {
      int col = colBase + u * 16 + l15;
      if (col >= N) col = N - 1;  // clamp for N=97 tail; store is guarded
      load_b_frag(W, K, col, k0, half, b[u]);
    }
#pragma unroll
    for (int t = 0; t < 2; ++t)
#pragma unroll
      for (int u = 0; u < 4; ++u)
        acc[t][u] = __builtin_amdgcn_wmma_f32_16x16x32_bf16(
            false, a[t], false, b[u], (short)0, acc[t][u], false, false);
  }
}

// ---------------------------------------------------------------------------
// Generic WMMA GEMM:  C[M,N] = A1[M,K1] @ B1[N,K1]^T (+ A2 @ B2^T) (+ biases)
// Block = 256 threads = 8 waves; block tile 128x128; wave tile 32x64.
// M must be a multiple of 128 (65536 / 256 / 6400 all are); N arbitrary.
// ---------------------------------------------------------------------------
__global__ __launch_bounds__(256) void wmma_gemm(
    const __bf16* __restrict__ A1, const __bf16* __restrict__ B1, int K1,
    const __bf16* __restrict__ A2, const __bf16* __restrict__ B2, int K2,
    const float* __restrict__ bias1, const float* __restrict__ bias2,
    float* __restrict__ Cf, __bf16* __restrict__ Cbf, int N) {
  const int lane = threadIdx.x & 31;
  const int wave = threadIdx.x >> 5;
  const int half = lane >> 4;
  const int l15 = lane & 15;
  const int wm = wave >> 1;  // 0..3
  const int wn = wave & 1;   // 0..1
  const int rowBase = blockIdx.y * 128 + wm * 32;
  const int colBase = blockIdx.x * 128 + wn * 64;

  v8f acc[2][4];
#pragma unroll
  for (int t = 0; t < 2; ++t)
#pragma unroll
    for (int u = 0; u < 4; ++u)
#pragma unroll
      for (int v = 0; v < 8; ++v) acc[t][u][v] = 0.0f;

  gemm_k_loop(A1, B1, K1, N, rowBase, colBase, half, l15, acc);
  if (K2 > 0) gemm_k_loop(A2, B2, K2, N, rowBase, colBase, half, l15, acc);

  // C/D layout: VGPR v -> row = rowBase + t*16 + half*8 + v, col = l15.
#pragma unroll
  for (int t = 0; t < 2; ++t)
#pragma unroll
    for (int u = 0; u < 4; ++u) {
      int col = colBase + u * 16 + l15;
      if (col < N) {
        float bias = 0.0f;
        if (bias1) bias += bias1[col];
        if (bias2) bias += bias2[col];
#pragma unroll
        for (int v = 0; v < 8; ++v) {
          int row = rowBase + t * 16 + half * 8 + v;
          float val = acc[t][u][v] + bias;
          size_t o = (size_t)row * N + col;
          if (Cf) Cf[o] = val;
          if (Cbf) Cbf[o] = (__bf16)val;
        }
      }
    }
}

// ---------------------------------------------------------------------------
// Fused attention step: one block (256 thr = 8 waves) per batch row b.
//   e_t   = sum_h tanh(Hproj[b,t,h] + ph[b,h]) * w[h]
//   alpha = softmax_t(e)          (also written to attention-map output)
//   x[b]  = [ sum_t alpha_t * bH[b,t,:] ,  emb[text[b,step]] ]   (bf16)
// ---------------------------------------------------------------------------
__global__ __launch_bounds__(256) void attn_step(
    const __bf16* __restrict__ hproj, const float* __restrict__ ph,
    const float* __restrict__ w_score, const __bf16* __restrict__ bh,
    const int* __restrict__ text, const float* __restrict__ emb, int step,
    __bf16* __restrict__ xbf, float* __restrict__ attn_out) {
  __shared__ float ph_s[Hc];
  __shared__ float w_s[Hc];
  __shared__ float e_s[Tc];
  __shared__ float red[Tc];

  const int b = blockIdx.x;
  const int tid = threadIdx.x;
  const int lane = tid & 31;
  const int wave = tid >> 5;

  for (int i = tid; i < Hc; i += 256) {
    ph_s[i] = ph[b * Hc + i];
    w_s[i] = w_score[i];
  }
  __syncthreads();

  // per-lane private chunk of h: [lane*16, lane*16+16)
  float phr[16], wr[16];
#pragma unroll
  for (int j = 0; j < 16; ++j) {
    phr[j] = ph_s[lane * 16 + j];
    wr[j] = w_s[lane * 16 + j];
  }

  // each wave scores 32 rows of t
  for (int it = 0; it < 32; ++it) {
    const int t = wave * 32 + it;
    const __bf16* hp = hproj + ((size_t)b * Tc + t) * Hc + lane * 16;
    v16bf hv;
    uint4* u = reinterpret_cast<uint4*>(&hv);
    u[0] = *reinterpret_cast<const uint4*>(hp);
    u[1] = *reinterpret_cast<const uint4*>(hp + 8);
    float partial = 0.0f;
#pragma unroll
    for (int j = 0; j < 16; ++j)
      partial += fast_tanh((float)hv[j] + phr[j]) * wr[j];
#pragma unroll
    for (int off = 16; off > 0; off >>= 1)
      partial += __shfl_xor(partial, off, 32);
    if (lane == 0) e_s[t] = partial;
  }
  __syncthreads();

  // softmax over T=256 (one element per thread)
  float e = e_s[tid];
  red[tid] = e;
  __syncthreads();
  for (int s = 128; s > 0; s >>= 1) {
    if (tid < s) red[tid] = fmaxf(red[tid], red[tid + s]);
    __syncthreads();
  }
  const float mx = red[0];
  __syncthreads();
  const float p = __expf(e - mx);
  red[tid] = p;
  __syncthreads();
  for (int s = 128; s > 0; s >>= 1) {
    if (tid < s) red[tid] += red[tid + s];
    __syncthreads();
  }
  const float alpha = p / red[0];
  e_s[tid] = alpha;
  attn_out[((size_t)b * Tc + tid) * STEPS + step] = alpha;  // [B,T,steps]
  __syncthreads();

  // context: x[b, i] = sum_t alpha_t * bH[b,t,i]
  for (int i = tid; i < INc; i += 256) {
    const __bf16* bp = bh + (size_t)b * Tc * INc + i;
    float acc = 0.0f;
    for (int t = 0; t < Tc; ++t) acc += e_s[t] * (float)bp[(size_t)t * INc];
    xbf[(size_t)b * (INc + Ec) + i] = (__bf16)acc;
  }
  // embedding half of x
  const int ch = text[b * 26 + step];
  for (int i = tid; i < Ec; i += 256)
    xbf[(size_t)b * (INc + Ec) + INc + i] = (__bf16)emb[(size_t)ch * Ec + i];
}

// ---------------------------------------------------------------------------
// LSTM pointwise: gates[B,4H] -> c,h ; h to f32 output + bf16 state copies
// ---------------------------------------------------------------------------
__global__ __launch_bounds__(256) void lstm_point(
    const float* __restrict__ gates, float* __restrict__ c,
    __bf16* __restrict__ h_bf, __bf16* __restrict__ hid_bf,
    float* __restrict__ hid_out, int step) {
  const int idx = blockIdx.x * 256 + threadIdx.x;  // B*H
  const int b = idx >> 9;
  const int j = idx & (Hc - 1);
  const float* g = gates + (size_t)b * 4 * Hc;
  const float ig = sigmoidf(g[j]);
  const float fg = sigmoidf(g[Hc + j]);
  const float gg = fast_tanh(g[2 * Hc + j]);
  const float og = sigmoidf(g[3 * Hc + j]);
  const float cn = fg * c[idx] + ig * gg;
  const float hn = og * fast_tanh(cn);
  c[idx] = cn;
  h_bf[idx] = (__bf16)hn;
  const size_t o = ((size_t)b * STEPS + step) * Hc + j;
  hid_bf[o] = (__bf16)hn;
  hid_out[o] = hn;
}

// ---------------------------------------------------------------------------
// f32 -> bf16 convert ; zero-init of (h, c) state
// ---------------------------------------------------------------------------
__global__ void cvt_bf16(const float* __restrict__ s, __bf16* __restrict__ d,
                         int n) {
  int i = blockIdx.x * 256 + threadIdx.x;
  if (i < n) d[i] = (__bf16)s[i];
}

__global__ void init_state(float* __restrict__ c, __bf16* __restrict__ h_bf,
                           int n) {
  int i = blockIdx.x * 256 + threadIdx.x;
  if (i < n) {
    c[i] = 0.0f;
    h_bf[i] = (__bf16)0.0f;
  }
}

// ---------------------------------------------------------------------------
// launch
// ---------------------------------------------------------------------------
extern "C" void kernel_launch(void* const* d_in, const int* in_sizes, int n_in,
                              void* d_out, int out_size, void* d_ws,
                              size_t ws_size, hipStream_t stream) {
  const float* batch_H = (const float*)d_in[0];
  const int* text = (const int*)d_in[1];
  // d_in[2] = batch_max_length (statically 25)
  const float* W_i2h = (const float*)d_in[3];
  const float* W_h2h = (const float*)d_in[4];
  const float* b_h2h = (const float*)d_in[5];
  const float* w_score = (const float*)d_in[6];
  const float* W_ih = (const float*)d_in[7];
  const float* W_hh = (const float*)d_in[8];
  const float* b_ih = (const float*)d_in[9];
  const float* b_hh = (const float*)d_in[10];
  const float* W_gen = (const float*)d_in[11];
  const float* b_gen = (const float*)d_in[12];
  const float* emb = (const float*)d_in[13];

  float* out = (float*)d_out;
  float* probs_out = out;                                    // [B,25,NC]
  float* attn_out = out + (size_t)Bc * STEPS * NCc;          // [B,T,25]
  float* hid_out = attn_out + (size_t)Bc * Tc * STEPS;       // [B,25,H]

  // workspace carve-up
  char* w = (char*)d_ws;
  auto alloc = [&](size_t bytes) -> void* {
    void* p = (void*)w;
    w += (bytes + 255) & ~(size_t)255;
    return p;
  };
  __bf16* bh_bf = (__bf16*)alloc((size_t)Bc * Tc * INc * 2);
  __bf16* hproj_bf = (__bf16*)alloc((size_t)Bc * Tc * Hc * 2);
  __bf16* wi2h_bf = (__bf16*)alloc((size_t)Hc * INc * 2);
  __bf16* wh2h_bf = (__bf16*)alloc((size_t)Hc * Hc * 2);
  __bf16* wih_bf = (__bf16*)alloc((size_t)4 * Hc * (INc + Ec) * 2);
  __bf16* whh_bf = (__bf16*)alloc((size_t)4 * Hc * Hc * 2);
  __bf16* wgen_bf = (__bf16*)alloc((size_t)NCc * Hc * 2);
  float* ph = (float*)alloc((size_t)Bc * Hc * 4);
  __bf16* xbf = (__bf16*)alloc((size_t)Bc * (INc + Ec) * 2);
  __bf16* h_bf = (__bf16*)alloc((size_t)Bc * Hc * 2);
  float* c_st = (float*)alloc((size_t)Bc * Hc * 4);
  float* gates = (float*)alloc((size_t)Bc * 4 * Hc * 4);
  __bf16* hid_bf = (__bf16*)alloc((size_t)Bc * STEPS * Hc * 2);

  auto cvt = [&](const float* s, __bf16* d, int n) {
    cvt_bf16<<<(n + 255) / 256, 256, 0, stream>>>(s, d, n);
  };
  cvt(batch_H, bh_bf, Bc * Tc * INc);
  cvt(W_i2h, wi2h_bf, Hc * INc);
  cvt(W_h2h, wh2h_bf, Hc * Hc);
  cvt(W_ih, wih_bf, 4 * Hc * (INc + Ec));
  cvt(W_hh, whh_bf, 4 * Hc * Hc);
  cvt(W_gen, wgen_bf, NCc * Hc);

  // H_proj = batch_H @ W_i2h^T  -> bf16 [B*T, H]
  wmma_gemm<<<dim3(Hc / 128, (Bc * Tc) / 128), 256, 0, stream>>>(
      bh_bf, wi2h_bf, INc, nullptr, nullptr, 0, nullptr, nullptr, nullptr,
      hproj_bf, Hc);

  init_state<<<(Bc * Hc + 255) / 256, 256, 0, stream>>>(c_st, h_bf, Bc * Hc);

  for (int step = 0; step < STEPS; ++step) {
    // ph = h @ W_h2h^T + b_h2h   [B,H]
    wmma_gemm<<<dim3(Hc / 128, Bc / 128), 256, 0, stream>>>(
        h_bf, wh2h_bf, Hc, nullptr, nullptr, 0, b_h2h, nullptr, ph, nullptr,
        Hc);
    // attention + context + embedding concat
    attn_step<<<Bc, 256, 0, stream>>>(hproj_bf, ph, w_score, bh_bf, text, emb,
                                      step, xbf, attn_out);
    // gates = x @ W_ih^T + h @ W_hh^T + b_ih + b_hh   [B,4H]
    wmma_gemm<<<dim3((4 * Hc) / 128, Bc / 128), 256, 0, stream>>>(
        xbf, wih_bf, INc + Ec, h_bf, whh_bf, Hc, b_ih, b_hh, gates, nullptr,
        4 * Hc);
    // LSTM pointwise
    lstm_point<<<(Bc * Hc) / 256, 256, 0, stream>>>(gates, c_st, h_bf, hid_bf,
                                                    hid_out, step);
  }

  // probs = hiddens @ W_gen^T + b_gen   [B*25, NC]
  wmma_gemm<<<dim3((NCc + 127) / 128, (Bc * STEPS) / 128), 256, 0, stream>>>(
      hid_bf, wgen_bf, Hc, nullptr, nullptr, 0, b_gen, nullptr, probs_out,
      nullptr, NCc);
}
